// S4sequence_27315992002810
// MI455X (gfx1250) — matrix operand order repaired and checked by reading, not compiled, not run on hardware
//
#include <hip/hip_runtime.h>
#include <hip/hip_bf16.h>
#include <math.h>

#define B_  16
#define L_  4096
#define H_  128
#define N_  64
#define Q_  64
#define NC_ 64   /* L_/Q_ chunks */

typedef __bf16 bf16;
typedef __attribute__((ext_vector_type(16))) __bf16 v16bf;
typedef __attribute__((ext_vector_type(8)))  float  v8f;

// ---------------- CDNA5 async global->LDS staging (ASYNCcnt tracked) ----------------
// GLOBAL_LOAD_ASYNC_TO_LDS_B128, GV mode: per-lane LDS byte offset in VDST VGPR,
// 64-bit global address in VADDR pair, saddr = off.
__device__ inline void async_ld_b128(void* lds_dst, const void* gsrc) {
  unsigned lds = (unsigned)(unsigned long long)lds_dst;  // low 32 bits of generic = LDS offset
  asm volatile("global_load_async_to_lds_b128 %0, %1, off"
               :: "v"(lds), "v"(gsrc) : "memory");
}
__device__ inline void async_wait0() {
  asm volatile("s_wait_asynccnt 0x0" ::: "memory");
}

// ---------------- WMMA fragment loaders (CDNA5 wave32 VGPR layouts) ----------------
// A-matrix 16x32 bf16: lanes 0-15 => M=lane, VGPR0-3 K=0..7, VGPR4-7 K=16..23;
// lanes 16-31 same +8 on K. Element (m,k) at p[m*rs + k*cs].
__device__ inline v16bf wmma_load_a(const bf16* __restrict__ p, int m0, int k0,
                                    int rs, int cs, int lane) {
  v16bf a;
  const int ml   = lane & 15;
  const int kadd = (lane >= 16) ? 8 : 0;
#pragma unroll
  for (int e = 0; e < 16; ++e) {
    int g = e >> 1, pos = e & 1;
    int k = (g < 4) ? (2 * g + pos) : (16 + 2 * (g - 4) + pos);
    k += kadd;
    a[e] = p[(size_t)(m0 + ml) * rs + (size_t)(k0 + k) * cs];
  }
  return a;
}
// B-matrix 32x16 bf16: lane%16 = N; lanes 0-15 hold K=0..7,16..23; lanes16-31 same +8 on K.
__device__ inline v16bf wmma_load_b(const bf16* __restrict__ p, int k0, int n0,
                                    int rs, int cs, int lane) {
  v16bf b;
  const int nl = lane & 15;
  const int kb = (lane >= 16) ? 8 : 0;
#pragma unroll
  for (int e = 0; e < 16; ++e)
    b[e] = p[(size_t)(k0 + kb + e + ((e >= 8) ? 8 : 0)) * rs + (size_t)(n0 + nl) * cs];
  return b;
}

// ---------------- K0: LayerNorm partial reduction ----------------
__global__ void k_ln_partial(const float* __restrict__ x, float* __restrict__ part) {
  const int b = blockIdx.x, s = blockIdx.y;          // 16 x 32
  const float* xb = x + (size_t)b * L_ * H_ + (size_t)s * 16384;
  float s1 = 0.f, s2 = 0.f;
  for (int i = threadIdx.x; i < 16384; i += 256) { float v = xb[i]; s1 += v; s2 += v * v; }
  __shared__ float r1[256], r2[256];
  r1[threadIdx.x] = s1; r2[threadIdx.x] = s2; __syncthreads();
  for (int o = 128; o > 0; o >>= 1) {
    if (threadIdx.x < o) { r1[threadIdx.x] += r1[threadIdx.x + o]; r2[threadIdx.x] += r2[threadIdx.x + o]; }
    __syncthreads();
  }
  if (threadIdx.x == 0) { part[(b * 32 + s) * 2 + 0] = r1[0]; part[(b * 32 + s) * 2 + 1] = r2[0]; }
}

// ---------------- K1: LayerNorm finalize (mean, rstd per batch) ----------------
__global__ void k_ln_final(const float* __restrict__ part, float* __restrict__ stats) {
  const int b = blockIdx.x;
  __shared__ float r1[32], r2[32];
  const int t = threadIdx.x;
  r1[t] = part[(b * 32 + t) * 2 + 0];
  r2[t] = part[(b * 32 + t) * 2 + 1];
  __syncthreads();
  for (int o = 16; o > 0; o >>= 1) {
    if (t < o) { r1[t] += r1[t + o]; r2[t] += r2[t + o]; }
    __syncthreads();
  }
  if (t == 0) {
    const float inv = 1.0f / (float)(L_ * H_);
    float m = r1[0] * inv;
    float var = r2[0] * inv - m * m;
    stats[b * 2 + 0] = m;
    stats[b * 2 + 1] = rsqrtf(var + 1e-5f);
  }
}

// ---------------- K2: normalize + transpose [B,L,H] -> bf16 [B,H,L] (LDS tiled) ----------------
__global__ void k_transpose_norm(const float* __restrict__ x, const float* __restrict__ stats,
                                 bf16* __restrict__ un) {
  const int b  = blockIdx.z;
  const int l0 = blockIdx.x * 32, h0 = blockIdx.y * 32;
  const int tx = threadIdx.x & 31, ty = threadIdx.x >> 5;   // ty in 0..7
  __shared__ float tile[32][33];
  const float mean = stats[b * 2 + 0], rstd = stats[b * 2 + 1];
  const float* xb = x + (size_t)b * L_ * H_;
#pragma unroll
  for (int r = 0; r < 4; ++r) {
    const int lr = ty + r * 8;
    tile[lr][tx] = xb[(size_t)(l0 + lr) * H_ + h0 + tx];    // coalesced over h
  }
  __syncthreads();
  bf16* ub = un + (size_t)b * H_ * L_;
#pragma unroll
  for (int r = 0; r < 4; ++r) {
    const int hr = ty + r * 8;
    ub[(size_t)(h0 + hr) * L_ + l0 + tx] = (bf16)((tile[tx][hr] - mean) * rstd); // coalesced over l
  }
}

// ---------------- K3: per-channel materials: Toeplitz T, M (enter->out), V (inj), w^Q, W1->bf16 ----
__global__ void k_materials(const float* __restrict__ A_re, const float* __restrict__ A_im,
                            const float* __restrict__ Bre,  const float* __restrict__ Bim,
                            const float* __restrict__ Cre,  const float* __restrict__ Cim,
                            const float* __restrict__ log_dt, const float* __restrict__ W1,
                            bf16* __restrict__ Tm,  bf16* __restrict__ Mre, bf16* __restrict__ MimN,
                            bf16* __restrict__ Vre, bf16* __restrict__ Vim,
                            float* __restrict__ wqre, float* __restrict__ wqim,
                            bf16* __restrict__ w1bf) {
  const int h = blockIdx.x, t = threadIdx.x;
  const float dt = __expf(log_dt[h]);
  __shared__ float Ksh[64];
  if (t < 64) {
    // K[d] = Re( sum_n CB_n * exp(dtA_n * d) ), d = t
    float kr = 0.f;
    const float d = (float)t;
    for (int n = 0; n < 64; ++n) {
      const int idx = h * 64 + n;
      const float ar = A_re[idx] * dt, ai = A_im[idx] * dt;
      const float br = Bre[idx], bi = Bim[idx], cr = Cre[idx], ci = Cim[idx];
      const float cbr = (cr * br - ci * bi) * dt;
      const float cbi = (cr * bi + ci * br) * dt;
      const float e = __expf(ar * d);
      float sn, cs; __sincosf(ai * d, &sn, &cs);
      kr += cbr * (e * cs) - cbi * (e * sn);
    }
    Ksh[t] = kr;
    // w^Q per state n = t
    const int idx = h * 64 + t;
    const float ar = A_re[idx] * dt, ai = A_im[idx] * dt;
    const float e = __expf(ar * 64.f);
    float sn, cs; __sincosf(ai * 64.f, &sn, &cs);
    wqre[idx] = e * cs;
    wqim[idx] = e * sn;
  }
  __syncthreads();
  // T[i][j] = K[i-j], lower triangular
  for (int idx = t; idx < 4096; idx += 256) {
    const int i = idx >> 6, j = idx & 63;
    Tm[(size_t)h * 4096 + idx] = (bf16)((i >= j) ? Ksh[i - j] : 0.f);
  }
  // M[i][n] = CB_n * w_n^(i+1): store Re and -Im (so both accumulate positively)
  for (int idx = t; idx < 4096; idx += 256) {
    const int i = idx >> 6, n = idx & 63;
    const int gi = h * 64 + n;
    const float ar = A_re[gi] * dt, ai = A_im[gi] * dt;
    const float br = Bre[gi], bi = Bim[gi], cr = Cre[gi], ci = Cim[gi];
    const float cbr = (cr * br - ci * bi) * dt;
    const float cbi = (cr * bi + ci * br) * dt;
    const float p = (float)(i + 1);
    const float e = __expf(ar * p);
    float sn, cs; __sincosf(ai * p, &sn, &cs);
    const float wr = e * cs, wi = e * sn;
    Mre [(size_t)h * 4096 + idx] = (bf16)(cbr * wr - cbi * wi);
    MimN[(size_t)h * 4096 + idx] = (bf16)(-(cbr * wi + cbi * wr));
  }
  // V[n][j] = w_n^(63-j)   (injection weights)
  for (int idx = t; idx < 4096; idx += 256) {
    const int n = idx >> 6, j = idx & 63;
    const int gi = h * 64 + n;
    const float ar = A_re[gi] * dt, ai = A_im[gi] * dt;
    const float p = (float)(63 - j);
    const float e = __expf(ar * p);
    float sn, cs; __sincosf(ai * p, &sn, &cs);
    Vre[(size_t)h * 4096 + idx] = (bf16)(e * cs);
    Vim[(size_t)h * 4096 + idx] = (bf16)(e * sn);
  }
  // W1 row h -> bf16
  if (t < 128) w1bf[h * 128 + t] = (bf16)W1[h * 128 + t];
}

// ---------------- K4: sequential chunk scan, stores entering state per chunk ----------------
__global__ void k_scan(const bf16* __restrict__ un, const bf16* __restrict__ Vre,
                       const bf16* __restrict__ Vim, const float* __restrict__ wqre,
                       const float* __restrict__ wqim, bf16* __restrict__ sre,
                       bf16* __restrict__ sim) {
  const int bh = blockIdx.x;            // b*128 + h
  const int h  = bh & 127;
  const int n  = threadIdx.x;           // state index, 64 threads (2 waves)
  const bf16* u  = un  + (size_t)bh * L_;
  const bf16* vrg = Vre + (size_t)h * 4096;
  const bf16* vig = Vim + (size_t)h * 4096;
  const float wr = wqre[h * 64 + n], wi = wqim[h * 64 + n];

  // Stage V matrices into LDS once via async global->LDS copies (8KB each).
  __shared__ bf16 vre_s[4096], vim_s[4096];
  __shared__ float uch[64];
#pragma unroll
  for (int i = 0; i < 8; ++i) {
    const int o = (n + i * 64) * 8;     // 512 b128 transfers per array
    async_ld_b128(&vre_s[o], vrg + o);
    async_ld_b128(&vim_s[o], vig + o);
  }
  async_wait0();
  __syncthreads();

  const bf16* vr = &vre_s[n * 64];
  const bf16* vi = &vim_s[n * 64];
  float sr = 0.f, si = 0.f;
  for (int c = 0; c < NC_; ++c) {
    uch[n] = (float)u[c * 64 + n];
    __syncthreads();
    sre[(size_t)bh * 4096 + c * 64 + n] = (bf16)sr;    // entering state of chunk c
    sim[(size_t)bh * 4096 + c * 64 + n] = (bf16)si;
    float ir = 0.f, ii = 0.f;
#pragma unroll 8
    for (int j = 0; j < 64; ++j) {
      const float uj = uch[j];
      ir += (float)vr[j] * uj;
      ii += (float)vi[j] * uj;
    }
    const float nsr = wr * sr - wi * si + ir;
    const float nsi = wr * si + wi * sr + ii;
    sr = nsr; si = nsi;
    __syncthreads();
  }
}

// ---------------- K5: Y = T@U + Mre@Sre + MimN@Sim (WMMA), + D*u, GELU -> g[B,L,H] bf16 -------
// All six 64x64 bf16 operand tiles staged in LDS via async copies (48KB/WG);
// each tile is otherwise read 4x redundantly by the 16 waves.
__global__ void k_s4combine(const bf16* __restrict__ un, const bf16* __restrict__ Tm,
                            const bf16* __restrict__ Mre, const bf16* __restrict__ MimN,
                            const bf16* __restrict__ sre, const bf16* __restrict__ sim,
                            const float* __restrict__ D, bf16* __restrict__ g) {
  const int bh = blockIdx.x;
  const int b = bh >> 7, h = bh & 127;
  const bf16* u  = un  + (size_t)bh * L_;
  const bf16* T  = Tm  + (size_t)h * 4096;
  const bf16* mr = Mre + (size_t)h * 4096;
  const bf16* mi = MimN+ (size_t)h * 4096;
  const bf16* xr = sre + (size_t)bh * 4096;
  const bf16* xi = sim + (size_t)bh * 4096;

  __shared__ bf16 sT[4096], sMr[4096], sMi[4096], sU[4096], sSr[4096], sSi[4096];
  const int tid = threadIdx.x;          // 512 threads: one b128 per thread per array
  {
    const int o = tid * 8;
    async_ld_b128(&sT [o], T  + o);
    async_ld_b128(&sMr[o], mr + o);
    async_ld_b128(&sMi[o], mi + o);
    async_ld_b128(&sU [o], u  + o);
    async_ld_b128(&sSr[o], xr + o);
    async_ld_b128(&sSi[o], xi + o);
  }
  async_wait0();
  __syncthreads();

  const int wave = tid >> 5, lane = tid & 31;
  const int mt = wave >> 2, nt = wave & 3;     // 16 waves -> 4x4 tiles of 16x16 over 64x64
  v8f c;
#pragma unroll
  for (int e = 0; e < 8; ++e) c[e] = 0.f;
#pragma unroll
  for (int k0 = 0; k0 < 64; k0 += 32) {
    // intra-chunk Toeplitz: (i,j) x U(j, chunk) ; U(j,c) = u[c*64+j]  (rs=1, cs=64)
    v16bf a  = wmma_load_a(sT, mt * 16, k0, 64, 1, lane);
    v16bf bb = wmma_load_b(sU, k0, nt * 16, 1, 64, lane);
    c = __builtin_amdgcn_wmma_f32_16x16x32_bf16(false, a, false, bb, (short)0, c, false, false);
    // entering state (real part): M_re(i,n) x Sre(n,c) ; Sre(n,c) = sre[c*64+n]
    a  = wmma_load_a(sMr, mt * 16, k0, 64, 1, lane);
    bb = wmma_load_b(sSr, k0, nt * 16, 1, 64, lane);
    c = __builtin_amdgcn_wmma_f32_16x16x32_bf16(false, a, false, bb, (short)0, c, false, false);
    // entering state (imag part, M_im pre-negated)
    a  = wmma_load_a(sMi, mt * 16, k0, 64, 1, lane);
    bb = wmma_load_b(sSi, k0, nt * 16, 1, 64, lane);
    c = __builtin_amdgcn_wmma_f32_16x16x32_bf16(false, a, false, bb, (short)0, c, false, false);
  }
  const float Dh = D[h];
  const int cc = nt * 16 + (lane & 15);        // chunk index (output column)
  const int rbase = (lane >= 16) ? 8 : 0;
#pragma unroll
  for (int r = 0; r < 8; ++r) {
    const int i = mt * 16 + rbase + r;         // position within chunk (output row)
    const int l = cc * 64 + i;
    const float uv = (float)sU[l];
    const float y  = c[r] + Dh * uv;
    const float ge = 0.5f * y * (1.f + erff(y * 0.70710678f));
    g[((size_t)b * L_ + l) * H_ + h] = (bf16)ge;
  }
}

// ---------------- K6: out = g @ W1^T + b1 + x   (65536 x 128 x 128, WMMA) ----------------
__global__ void k_out_gemm(const bf16* __restrict__ g, const bf16* __restrict__ w1bf,
                           const float* __restrict__ b1, const float* __restrict__ x,
                           float* __restrict__ out) {
  const int row0 = blockIdx.x * 64;
  const int col0 = blockIdx.y * 64;
  const int wave = threadIdx.x >> 5, lane = threadIdx.x & 31;
  const int mt = wave >> 2, nt = wave & 3;
  v8f c;
#pragma unroll
  for (int e = 0; e < 8; ++e) c[e] = 0.f;
#pragma unroll
  for (int k0 = 0; k0 < 128; k0 += 32) {
    v16bf a  = wmma_load_a(g + (size_t)row0 * H_, mt * 16, k0, H_, 1, lane);
    // B(k=h, n=o) = W1[o*128 + h]  => rs=1, cs=128
    v16bf bb = wmma_load_b(w1bf, k0, col0 + nt * 16, 1, H_, lane);
    c = __builtin_amdgcn_wmma_f32_16x16x32_bf16(false, a, false, bb, (short)0, c, false, false);
  }
  const int n = col0 + nt * 16 + (lane & 15);
  const float bias = b1[n];
  const int rbase = (lane >= 16) ? 8 : 0;
#pragma unroll
  for (int r = 0; r < 8; ++r) {
    const int m = row0 + mt * 16 + rbase + r;  // m = b*L + l
    out[(size_t)m * H_ + n] = c[r] + bias + x[(size_t)m * H_ + n];
  }
}

// ---------------- launcher ----------------
extern "C" void kernel_launch(void* const* d_in, const int* in_sizes, int n_in,
                              void* d_out, int out_size, void* d_ws, size_t ws_size,
                              hipStream_t stream) {
  const float* x      = (const float*)d_in[0];
  const float* A_re   = (const float*)d_in[1];
  const float* A_im   = (const float*)d_in[2];
  const float* Bre    = (const float*)d_in[3];
  const float* Bim    = (const float*)d_in[4];
  const float* Cre    = (const float*)d_in[5];
  const float* Cim    = (const float*)d_in[6];
  const float* D      = (const float*)d_in[7];
  const float* log_dt = (const float*)d_in[8];
  const float* W1     = (const float*)d_in[9];
  const float* b1     = (const float*)d_in[10];
  float* out = (float*)d_out;
  (void)in_sizes; (void)n_in; (void)out_size; (void)ws_size;

  char* w = (char*)d_ws;
  auto take = [&](size_t bytes) -> char* {
    char* p = w;
    w += (bytes + 255) & ~(size_t)255;
    return p;
  };
  float* part  = (float*)take(16 * 32 * 2 * sizeof(float));
  float* stats = (float*)take(16 * 2 * sizeof(float));
  bf16*  un    = (bf16*) take((size_t)B_ * H_ * L_ * sizeof(bf16));    // 16 MB
  bf16*  Tm    = (bf16*) take((size_t)H_ * 4096 * sizeof(bf16));       // 1 MB
  bf16*  Mre   = (bf16*) take((size_t)H_ * 4096 * sizeof(bf16));
  bf16*  MimN  = (bf16*) take((size_t)H_ * 4096 * sizeof(bf16));
  bf16*  Vre   = (bf16*) take((size_t)H_ * 4096 * sizeof(bf16));
  bf16*  Vim   = (bf16*) take((size_t)H_ * 4096 * sizeof(bf16));
  float* wqre  = (float*)take((size_t)H_ * 64 * sizeof(float));
  float* wqim  = (float*)take((size_t)H_ * 64 * sizeof(float));
  bf16*  sre   = (bf16*) take((size_t)B_ * H_ * 4096 * sizeof(bf16));  // 16 MB
  bf16*  sim   = (bf16*) take((size_t)B_ * H_ * 4096 * sizeof(bf16));  // 16 MB
  bf16*  gb    = (bf16*) take((size_t)B_ * L_ * H_ * sizeof(bf16));    // 16 MB
  bf16*  w1bf  = (bf16*) take((size_t)H_ * H_ * sizeof(bf16));

  k_ln_partial   <<<dim3(16, 32), 256, 0, stream>>>(x, part);
  k_ln_final     <<<16, 32, 0, stream>>>(part, stats);
  k_transpose_norm<<<dim3(L_ / 32, H_ / 32, B_), 256, 0, stream>>>(x, stats, un);
  k_materials    <<<H_, 256, 0, stream>>>(A_re, A_im, Bre, Bim, Cre, Cim, log_dt, W1,
                                          Tm, Mre, MimN, Vre, Vim, wqre, wqim, w1bf);
  k_scan         <<<B_ * H_, 64, 0, stream>>>(un, Vre, Vim, wqre, wqim, sre, sim);
  k_s4combine    <<<B_ * H_, 512, 0, stream>>>(un, Tm, Mre, MimN, sre, sim, D, gb);
  k_out_gemm     <<<dim3((B_ * L_) / 64, H_ / 64), 512, 0, stream>>>(gb, w1bf, b1, x, out);
}
